// Critic_13967233647660
// MI455X (gfx1250) — matrix operand order repaired and checked by reading, not compiled, not run on hardware
//
#include <hip/hip_runtime.h>
#include <hip/hip_bf16.h>

typedef _Float16 f16;
typedef _Float16 v16h __attribute__((ext_vector_type(16)));
typedef float    v8f  __attribute__((ext_vector_type(8)));

#define NA    8
#define NB    32768
#define NOBS  128
#define NACT  32
#define NE    128
#define NH    4
#define ND    32
#define CIN   160      // OBS + ACT
#define F1IN  256      // 2*E
#define CHUNK 4096
#define NCHUNK (NB / CHUNK)

union AFrag { v16h v; unsigned int u[8]; f16 h[16]; };
union HPack { f16 h[2]; unsigned int u; };

__device__ __forceinline__ float lrelu(float x) { return x > 0.f ? x : 0.01f * x; }

// A-fragment (16x32 f16): lane L -> row M=L&15, K-quarter L>>4.
// v0..3 hold K = kq*8 + {0..7} pairs, v4..7 hold K = 16 + kq*8 + {0..7} pairs.
__device__ __forceinline__ v16h load_a(const f16* base, int ld, int k0, int lane) {
  const int m = lane & 15, kq = lane >> 4;
  const f16* p = base + m * ld + k0 + kq * 8;
  AFrag f;
#pragma unroll
  for (int i = 0; i < 4; ++i) f.u[i] = *(const unsigned int*)(p + 2 * i);
#pragma unroll
  for (int i = 0; i < 4; ++i) f.u[4 + i] = *(const unsigned int*)(p + 16 + 2 * i);
  return f.v;
}

// B-fragment (32x16 f16) from transposed weights wt[n][k] (ld = K stride):
// lane L -> col N=L&15, K-quarter L>>4; v0..3 = K kq*8+{0..7}, v4..7 = K 16+kq*8+{0..7}.
__device__ __forceinline__ v16h load_b(const f16* wt, int ld, int k0, int lane) {
  const int n = lane & 15, kh = lane >> 4;
  const f16* p = wt + n * ld + k0 + kh * 8;
  AFrag f;
#pragma unroll
  for (int i = 0; i < 4; ++i) f.u[i] = *(const unsigned int*)(p + 2 * i);
#pragma unroll
  for (int i = 0; i < 4; ++i) f.u[4 + i] = *(const unsigned int*)(p + 16 + 2 * i);
  return f.v;
}

// ---------------- Kernel 1: BatchNorm statistics -> scale/shift -----------------
__global__ void __launch_bounds__(256) bn_stats_kernel(
    const float* __restrict__ obs, const float* __restrict__ act,
    const float* __restrict__ g_gamma, const float* __restrict__ g_beta,
    const float* __restrict__ s_gamma, const float* __restrict__ s_beta,
    float* __restrict__ g_scale, float* __restrict__ g_shift,
    float* __restrict__ s_scale, float* __restrict__ s_shift) {
  __shared__ float ssum[256], ssq[256];
  const int a = blockIdx.x / CIN;
  const int f = blockIdx.x % CIN;
  const float* src;
  int stride;
  if (f < NOBS) { src = obs + (size_t)a * NB * NOBS + f; stride = NOBS; }
  else          { src = act + (size_t)a * NB * NACT + (f - NOBS); stride = NACT; }
  float s = 0.f, s2 = 0.f;
  for (int b = threadIdx.x; b < NB; b += 256) {
    float x = src[(size_t)b * stride];
    s += x; s2 += x * x;
  }
  ssum[threadIdx.x] = s; ssq[threadIdx.x] = s2;
  __syncthreads();
  for (int w = 128; w > 0; w >>= 1) {
    if (threadIdx.x < w) {
      ssum[threadIdx.x] += ssum[threadIdx.x + w];
      ssq[threadIdx.x]  += ssq[threadIdx.x + w];
    }
    __syncthreads();
  }
  if (threadIdx.x == 0) {
    float mean = ssum[0] / (float)NB;
    float var  = ssq[0] / (float)NB - mean * mean;   // biased variance (torch BN fwd)
    float rstd = rsqrtf(var + 1e-5f);
    float gs = g_gamma[a * CIN + f] * rstd;
    g_scale[a * CIN + f] = gs;
    g_shift[a * CIN + f] = g_beta[a * CIN + f] - mean * gs;
    if (f < NOBS) {  // obs stats are shared between g-BN and s-BN
      float ss = s_gamma[a * NOBS + f] * rstd;
      s_scale[a * NOBS + f] = ss;
      s_shift[a * NOBS + f] = s_beta[a * NOBS + f] - mean * ss;
    }
  }
}

// ------------- Kernel 2: transpose fp32 weights -> f16 [batch][out][in] ---------
__global__ void pack_wT_kernel(const float* __restrict__ src, f16* __restrict__ dst,
                               int batch, int in, int out) {
  int total = batch * in * out;
  for (int idx = blockIdx.x * blockDim.x + threadIdx.x; idx < total;
       idx += gridDim.x * blockDim.x) {
    int b = idx / (in * out);
    int r = idx - b * in * out;
    int i = r / out, o = r - i * out;
    dst[(size_t)b * in * out + (size_t)o * in + i] = (f16)src[idx];
  }
}

// --- Kernel 3: per agent: BN-apply + e GEMM (LDS) + s GEMM + q/k/v GEMMs (WMMA) ---
__global__ void __launch_bounds__(256) es_qkv_kernel(
    const float* __restrict__ obs, const float* __restrict__ act,
    const float* __restrict__ g_scale, const float* __restrict__ g_shift,
    const float* __restrict__ s_scale, const float* __restrict__ s_shift,
    const f16* __restrict__ WgT, const f16* __restrict__ WsT, const f16* __restrict__ WqkvT,
    const float* __restrict__ bg, const float* __restrict__ bs,
    f16* __restrict__ qbuf, f16* __restrict__ kbuf, f16* __restrict__ vbuf,
    f16* __restrict__ sbuf, int chunkBase) {
  __shared__ alignas(16) f16 comb[4][16 * 162];
  __shared__ alignas(16) f16 sobs[4][16 * 130];
  __shared__ alignas(16) f16 etile[4][16 * 130];
  const int a    = blockIdx.y;
  const int wave = threadIdx.x >> 5;
  const int lane = threadIdx.x & 31;
  const int rowL = blockIdx.x * 64 + wave * 16;   // chunk-relative row base
  const size_t rowG = (size_t)chunkBase + rowL;   // absolute batch row base

  // stage normalized inputs as f16
  for (int idx = lane; idx < 16 * CIN; idx += 32) {
    int m = idx / CIN, f = idx - m * CIN;
    float x = (f < NOBS) ? obs[((size_t)a * NB + rowG + m) * NOBS + f]
                         : act[((size_t)a * NB + rowG + m) * NACT + (f - NOBS)];
    comb[wave][m * 162 + f] = (f16)(x * g_scale[a * CIN + f] + g_shift[a * CIN + f]);
  }
  for (int idx = lane; idx < 16 * NOBS; idx += 32) {
    int m = idx / NOBS, f = idx - m * NOBS;
    float x = obs[((size_t)a * NB + rowG + m) * NOBS + f];
    sobs[wave][m * 130 + f] = (f16)(x * s_scale[a * NOBS + f] + s_shift[a * NOBS + f]);
  }
  __syncthreads();

  const int cn = lane & 15;
  const int rr = lane >> 4;

  // e = lrelu(comb_n @ Wg + bg) -> LDS etile
#pragma unroll
  for (int nt = 0; nt < 8; ++nt) {
    v8f acc;
    float bias = bg[a * NE + nt * 16 + cn];
#pragma unroll
    for (int i = 0; i < 8; ++i) acc[i] = bias;
    const f16* wb = WgT + (size_t)a * NE * CIN + (size_t)(nt * 16) * CIN;
#pragma unroll
    for (int kk = 0; kk < CIN; kk += 32) {
      v16h af = load_a(comb[wave], 162, kk, lane);
      v16h bf = load_b(wb, CIN, kk, lane);
      acc = __builtin_amdgcn_wmma_f32_16x16x32_f16(false, af, false, bf, (short)0, acc,
                                                   false, false);
    }
#pragma unroll
    for (int i = 0; i < 8; ++i)
      etile[wave][(rr * 8 + i) * 130 + nt * 16 + cn] = (f16)lrelu(acc[i]);
  }
  __syncthreads();

  // s = lrelu(obs_n @ Ws + bs) -> global
#pragma unroll
  for (int nt = 0; nt < 8; ++nt) {
    v8f acc;
    float bias = bs[a * NE + nt * 16 + cn];
#pragma unroll
    for (int i = 0; i < 8; ++i) acc[i] = bias;
    const f16* wb = WsT + (size_t)a * NE * NOBS + (size_t)(nt * 16) * NOBS;
#pragma unroll
    for (int kk = 0; kk < NOBS; kk += 32) {
      v16h af = load_a(sobs[wave], 130, kk, lane);
      v16h bf = load_b(wb, NOBS, kk, lane);
      acc = __builtin_amdgcn_wmma_f32_16x16x32_f16(false, af, false, bf, (short)0, acc,
                                                   false, false);
    }
#pragma unroll
    for (int i = 0; i < 8; ++i)
      sbuf[((size_t)a * CHUNK + rowL + rr * 8 + i) * NE + nt * 16 + cn] = (f16)lrelu(acc[i]);
  }

  // q,k,v = e @ W{q,k,v} (head-packed columns h*32+d); v gets lrelu
  f16* outs[3] = {qbuf, kbuf, vbuf};
#pragma unroll
  for (int w = 0; w < 3; ++w) {
#pragma unroll
    for (int nt = 0; nt < 8; ++nt) {
      v8f acc;
#pragma unroll
      for (int i = 0; i < 8; ++i) acc[i] = 0.f;
      const f16* wb = WqkvT + (size_t)w * NE * NE + (size_t)(nt * 16) * NE;
#pragma unroll
      for (int kk = 0; kk < NE; kk += 32) {
        v16h af = load_a(etile[wave], 130, kk, lane);
        v16h bf = load_b(wb, NE, kk, lane);
        acc = __builtin_amdgcn_wmma_f32_16x16x32_f16(false, af, false, bf, (short)0, acc,
                                                     false, false);
      }
#pragma unroll
      for (int i = 0; i < 8; ++i) {
        float v = acc[i];
        if (w == 2) v = lrelu(v);
        outs[w][((size_t)a * CHUNK + rowL + rr * 8 + i) * NE + nt * 16 + cn] = (f16)v;
      }
    }
  }
}

// ---- Kernel 4: cross-agent attention (8x8 per (b,h), diag-masked, no softmax) ----
__global__ void __launch_bounds__(256) attn_kernel(
    const f16* __restrict__ qbuf, const f16* __restrict__ kbuf,
    const f16* __restrict__ vbuf, f16* __restrict__ xibuf) {
  __shared__ alignas(16) f16 qs[NA][8][NE];
  __shared__ alignas(16) f16 ks[NA][8][NE];
  __shared__ alignas(16) f16 vs[NA][8][NE];
  const int r0  = blockIdx.x * 8;   // chunk-relative
  const int tid = threadIdx.x;
  for (int idx = tid; idx < NA * 8 * (NE / 2); idx += 256) {
    int ag  = idx / (8 * 64);
    int rem = idx - ag * 8 * 64;
    int r = rem >> 6, fp = rem & 63;
    size_t g = ((size_t)ag * CHUNK + r0 + r) * NE + 2 * fp;
    *(unsigned int*)&qs[ag][r][2 * fp] = *(const unsigned int*)(qbuf + g);
    *(unsigned int*)&ks[ag][r][2 * fp] = *(const unsigned int*)(kbuf + g);
    *(unsigned int*)&vs[ag][r][2 * fp] = *(const unsigned int*)(vbuf + g);
  }
  __syncthreads();
  const int b = tid & 7;
  const int h = (tid >> 3) & 3;
  const int i = tid >> 5;
  float qr[ND], accv[ND];
#pragma unroll
  for (int d = 0; d < ND; ++d) { qr[d] = (float)qs[i][b][h * ND + d]; accv[d] = 0.f; }
  const float inv = 0.17677669529663687f;  // 1/sqrt(D), D=32
#pragma unroll
  for (int j = 0; j < NA; ++j) {
    if (j == i) continue;  // mask = 1 - eye(A)
    float al = 0.f;
#pragma unroll
    for (int d = 0; d < ND; ++d) al += qr[d] * (float)ks[j][b][h * ND + d];
    al *= inv;
#pragma unroll
    for (int d = 0; d < ND; ++d) accv[d] += al * (float)vs[j][b][h * ND + d];
  }
  size_t base = ((size_t)i * CHUNK + r0 + b) * NE + h * ND;
#pragma unroll
  for (int t = 0; t < ND / 2; ++t) {
    HPack p; p.h[0] = (f16)accv[2 * t]; p.h[1] = (f16)accv[2 * t + 1];
    *(unsigned int*)(xibuf + base + 2 * t) = p.u;
  }
}

// -------- Kernel 5: f-net (WMMA) + argmax(action) gather -> output ---------------
__global__ void __launch_bounds__(256) fnet_kernel(
    const f16* __restrict__ xibuf, const f16* __restrict__ sbuf,
    const f16* __restrict__ Wf1T, const f16* __restrict__ Wf2T,
    const float* __restrict__ bf1, const float* __restrict__ bf2,
    const float* __restrict__ act, float* __restrict__ out, int chunkBase) {
  __shared__ alignas(16) f16 h1[8][16 * 130];
  __shared__ alignas(16) float allq[8][16][NACT];
  const int a    = blockIdx.y;
  const int wave = threadIdx.x >> 5;
  const int lane = threadIdx.x & 31;
  const int rowL = blockIdx.x * 128 + wave * 16;  // chunk-relative
  const int cn = lane & 15, rr = lane >> 4;

  const f16* xiRow = xibuf + ((size_t)a * CHUNK + rowL) * NE;
  const f16* sRow  = sbuf  + ((size_t)a * CHUNK + rowL) * NE;

  // h1 = lrelu([xi, s] @ Wf1 + bf1)
#pragma unroll
  for (int nt = 0; nt < 8; ++nt) {
    v8f acc;
    float bias = bf1[a * NE + nt * 16 + cn];
#pragma unroll
    for (int i = 0; i < 8; ++i) acc[i] = bias;
    const f16* wb = Wf1T + (size_t)a * NE * F1IN + (size_t)(nt * 16) * F1IN;
#pragma unroll
    for (int kk = 0; kk < F1IN; kk += 32) {
      v16h af = (kk < NE) ? load_a(xiRow, NE, kk, lane)
                          : load_a(sRow, NE, kk - NE, lane);
      v16h bf = load_b(wb, F1IN, kk, lane);
      acc = __builtin_amdgcn_wmma_f32_16x16x32_f16(false, af, false, bf, (short)0, acc,
                                                   false, false);
    }
#pragma unroll
    for (int i = 0; i < 8; ++i)
      h1[wave][(rr * 8 + i) * 130 + nt * 16 + cn] = (f16)lrelu(acc[i]);
  }
  __syncthreads();

  // all_action_q = h1 @ Wf2 + bf2
#pragma unroll
  for (int nt = 0; nt < 2; ++nt) {
    v8f acc;
    float bias = bf2[a * NACT + nt * 16 + cn];
#pragma unroll
    for (int i = 0; i < 8; ++i) acc[i] = bias;
    const f16* wb = Wf2T + (size_t)a * NACT * NE + (size_t)(nt * 16) * NE;
#pragma unroll
    for (int kk = 0; kk < NE; kk += 32) {
      v16h af = load_a(h1[wave], 130, kk, lane);
      v16h bf = load_b(wb, NE, kk, lane);
      acc = __builtin_amdgcn_wmma_f32_16x16x32_f16(false, af, false, bf, (short)0, acc,
                                                   false, false);
    }
#pragma unroll
    for (int i = 0; i < 8; ++i)
      allq[wave][rr * 8 + i][nt * 16 + cn] = acc[i];
  }
  __syncthreads();

  // argmax over action_vector (first-max, like jnp.argmax) + gather
  if (lane < 16) {
    size_t brow = (size_t)chunkBase + rowL + lane;
    const float* av = act + ((size_t)a * NB + brow) * NACT;
    float best = av[0]; int bi = 0;
#pragma unroll
    for (int c = 1; c < NACT; ++c) {
      float v = av[c];
      if (v > best) { best = v; bi = c; }
    }
    out[(size_t)a * NB + brow] = allq[wave][lane][bi];
  }
}

// --------------------------------- host launcher ---------------------------------
extern "C" void kernel_launch(void* const* d_in, const int* in_sizes, int n_in,
                              void* d_out, int out_size, void* d_ws, size_t ws_size,
                              hipStream_t stream) {
  (void)in_sizes; (void)n_in; (void)out_size; (void)ws_size;
  const float* obs     = (const float*)d_in[0];
  const float* act     = (const float*)d_in[1];
  const float* g_gamma = (const float*)d_in[2];
  const float* g_beta  = (const float*)d_in[3];
  const float* Wg      = (const float*)d_in[4];
  const float* bg      = (const float*)d_in[5];
  const float* s_gamma = (const float*)d_in[6];
  const float* s_beta  = (const float*)d_in[7];
  const float* Ws      = (const float*)d_in[8];
  const float* bs      = (const float*)d_in[9];
  const float* Wq      = (const float*)d_in[10];
  const float* Wk      = (const float*)d_in[11];
  const float* Wv      = (const float*)d_in[12];
  const float* Wf1     = (const float*)d_in[13];
  const float* bf1     = (const float*)d_in[14];
  const float* Wf2     = (const float*)d_in[15];
  const float* bf2     = (const float*)d_in[16];
  float* out = (float*)d_out;

  char* ws = (char*)d_ws;
  size_t off = 0;
  auto alloc = [&](size_t bytes) {
    size_t r = off;
    off = (off + bytes + 255) & ~(size_t)255;
    return r;
  };
  float* g_scale = (float*)(ws + alloc((size_t)NA * CIN * 4));
  float* g_shift = (float*)(ws + alloc((size_t)NA * CIN * 4));
  float* s_scale = (float*)(ws + alloc((size_t)NA * NOBS * 4));
  float* s_shift = (float*)(ws + alloc((size_t)NA * NOBS * 4));
  f16* WgT   = (f16*)(ws + alloc((size_t)NA * NE * CIN * 2));
  f16* WsT   = (f16*)(ws + alloc((size_t)NA * NE * NOBS * 2));
  f16* WqkvT = (f16*)(ws + alloc((size_t)3 * NE * NE * 2));
  f16* Wf1T  = (f16*)(ws + alloc((size_t)NA * NE * F1IN * 2));
  f16* Wf2T  = (f16*)(ws + alloc((size_t)NA * NACT * NE * 2));
  f16* qbuf  = (f16*)(ws + alloc((size_t)NA * CHUNK * NE * 2));
  f16* kbuf  = (f16*)(ws + alloc((size_t)NA * CHUNK * NE * 2));
  f16* vbuf  = (f16*)(ws + alloc((size_t)NA * CHUNK * NE * 2));
  f16* sbuf  = (f16*)(ws + alloc((size_t)NA * CHUNK * NE * 2));
  f16* xibuf = (f16*)(ws + alloc((size_t)NA * CHUNK * NE * 2));

  bn_stats_kernel<<<NA * CIN, 256, 0, stream>>>(obs, act, g_gamma, g_beta, s_gamma, s_beta,
                                                g_scale, g_shift, s_scale, s_shift);

  pack_wT_kernel<<<640, 256, 0, stream>>>(Wg, WgT, NA, CIN, NE);
  pack_wT_kernel<<<512, 256, 0, stream>>>(Ws, WsT, NA, NOBS, NE);
  pack_wT_kernel<<<64, 256, 0, stream>>>(Wq, WqkvT, NH, NE, ND);
  pack_wT_kernel<<<64, 256, 0, stream>>>(Wk, WqkvT + (size_t)NE * NE, NH, NE, ND);
  pack_wT_kernel<<<64, 256, 0, stream>>>(Wv, WqkvT + (size_t)2 * NE * NE, NH, NE, ND);
  pack_wT_kernel<<<1024, 256, 0, stream>>>(Wf1, Wf1T, NA, F1IN, NE);
  pack_wT_kernel<<<128, 256, 0, stream>>>(Wf2, Wf2T, NA, NE, NACT);

  for (int c = 0; c < NCHUNK; ++c) {
    int cb = c * CHUNK;
    es_qkv_kernel<<<dim3(CHUNK / 64, NA), 256, 0, stream>>>(
        obs, act, g_scale, g_shift, s_scale, s_shift, WgT, WsT, WqkvT, bg, bs,
        qbuf, kbuf, vbuf, sbuf, cb);
    attn_kernel<<<CHUNK / 8, 256, 0, stream>>>(qbuf, kbuf, vbuf, xibuf);
    fnet_kernel<<<dim3(CHUNK / 128, NA), 256, 0, stream>>>(
        xibuf, sbuf, Wf1T, Wf2T, bf1, bf2, act, out, cb);
  }
}
